// ATN_59133109731772
// MI455X (gfx1250) — compile-verified, matrix operand order checked by reading, not run on hardware
//
#include <hip/hip_runtime.h>
#include <hip/hip_bf16.h>

typedef _Float16 f16;
typedef __attribute__((ext_vector_type(16))) _Float16 v16h;
typedef __attribute__((ext_vector_type(8)))  float    v8f;

#define C_IN   192
#define MIDC    96
#define HIMG   256
#define PADS     4
#define HP     264            // 256 + 2*4
#define NWSIDE  33            // 264/8
#define NWINB  (NWSIDE*NWSIDE) // 1089
#define NHEADS   8
#define CPH     12
#define EPSBN  1e-5f

// ---------------- fragment helpers ----------------

__device__ inline v16h frag_load(const f16* p) {
    // p must be 8-byte aligned (half-index % 4 == 0) -> 4x b64 loads
    union { v16h v; uint2 u[4]; } f;
    const uint2* q = reinterpret_cast<const uint2*>(p);
#pragma unroll
    for (int j = 0; j < 4; ++j) f.u[j] = q[j];
    return f.v;
}

__device__ inline v16h frag_zero() {
    union { v16h v; uint2 u[4]; } f;
#pragma unroll
    for (int j = 0; j < 4; ++j) f.u[j] = make_uint2(0u, 0u);
    return f.v;
}

// zero out K slots >= 12 (head-dim zero padding to K=32). Lane group lgr=0
// holds k = 0..15, lgr=1 holds k = 16..31 (all padding).
__device__ inline v16h mask_headdim(v16h v, int lgr) {
    union { v16h v; f16 h[16]; } f; f.v = v;
    if (lgr) {
#pragma unroll
        for (int j = 0; j < 16; ++j) f.h[j] = (f16)0.f;
    } else {
#pragma unroll
        for (int j = 12; j < 16; ++j) f.h[j] = (f16)0.f;
    }
    return f.v;
}

__device__ inline v8f wmma_f16(v16h a, v16h b, v8f c) {
    return __builtin_amdgcn_wmma_f32_16x16x32_f16(false, a, false, b,
                                                  (short)0, c, false, false);
}

// ---------------- kernel 0: fold BN + shift-mask into fp16 GEMM weights -----

struct FoldArgs {
    const float* w[3];      // theta, phi, g : (96,192,3,3)
    const float* b[3];
    const float* gamma[3];
    const float* beta[3];
    const float* mean[3];
    const float* var[3];
    const float* w_out;     // (192,96,3,3)
};

__global__ __launch_bounds__(256)
void fold_weights(FoldArgs fa, f16* wf_qkv, float* bf_qkv, f16* wf_out) {
    const int total_qkv  = 3 * MIDC * C_IN;   // 55296
    const int total_out  = C_IN * MIDC;       // 18432
    const int total_bias = 3 * MIDC;          // 288
    int stride = gridDim.x * blockDim.x;
    for (int e = blockIdx.x * blockDim.x + threadIdx.x;
         e < total_qkv + total_out + total_bias; e += stride) {
        if (e < total_qkv) {
            int conv = e / (MIDC * C_IN);
            int r    = e % (MIDC * C_IN);
            int o = r / C_IN, i = r % C_IN;
            int grp = i / 38; if (grp > 4) grp = 4;
            int kh = (grp == 2) ? 2 : ((grp == 3) ? 0 : 1);
            int kw = (grp == 0) ? 2 : ((grp == 1) ? 0 : 1);
            float scale = fa.gamma[conv][o] * rsqrtf(fa.var[conv][o] + EPSBN);
            float w = fa.w[conv][(o * C_IN + i) * 9 + kh * 3 + kw] * scale;
            wf_qkv[e] = (f16)w;
        } else if (e < total_qkv + total_out) {
            int r = e - total_qkv;
            int o = r / MIDC, i = r % MIDC;
            int grp = i / 19; if (grp > 4) grp = 4;
            int kh = (grp == 2) ? 2 : ((grp == 3) ? 0 : 1);
            int kw = (grp == 0) ? 2 : ((grp == 1) ? 0 : 1);
            wf_out[r] = (f16)fa.w_out[(o * MIDC + i) * 9 + kh * 3 + kw];
        } else {
            int r = e - total_qkv - total_out;
            int conv = r / MIDC, o = r % MIDC;
            float scale = fa.gamma[conv][o] * rsqrtf(fa.var[conv][o] + EPSBN);
            bf_qkv[r] = (fa.b[conv][o] - fa.mean[conv][o]) * scale + fa.beta[conv][o];
        }
    }
}

// ---------------- kernel 1: fused QKV-GEMM + window attention ---------------
// One workgroup (256 threads = 8 waves) per 8x8 window of the padded image.

__global__ __launch_bounds__(256)
void window_attn(const float* __restrict__ x,       // (4,192,256,256)
                 const f16*  __restrict__ wf_qkv,   // 3 x (96,192) fp16, BN-folded
                 const float* __restrict__ bf_qkv,  // 3 x 96
                 f16* __restrict__ attn_out)        // (4,96,264,264) fp16
{
    __shared__ __align__(16) f16 sh_theta[64 * MIDC + 32];  // [pix][ch] + overrun pad
    __shared__ __align__(16) f16 sh_phi  [64 * MIDC + 32];  // [pix][ch] + overrun pad
    __shared__ __align__(16) f16 sh_v    [MIDC * 64];       // [ch][pix]
    __shared__ __align__(16) f16 sh_scr  [64 * C_IN];       // B-matrix, later sim[2][64][64]
    __shared__ __align__(16) f16 sh_stage[24 * 64];         // staged attn output (2 heads)

    const int tid  = threadIdx.x;
    const int wv   = tid >> 5;
    const int lane = tid & 31;
    const int lrow = lane & 15;
    const int lgr  = lane >> 4;

    const int win = blockIdx.x;
    const int n   = win / NWINB;
    const int rw  = win % NWINB;
    const int y0  = (rw / NWSIDE) * 8;   // padded-image window origin
    const int x0  = (rw % NWSIDE) * 8;

    // warm the fp16 weight stream into cache while we gather x
    __builtin_prefetch(wf_qkv + (tid << 5), 0, 3);

    // ---- stage 1a: gather shifted input taps as B-matrix [pix][k] fp16 ----
    for (int idx = tid; idx < 64 * C_IN; idx += 256) {
        int pix = idx & 63;
        int i   = idx >> 6;
        int grp = i / 38; if (grp > 4) grp = 4;
        int dh = (grp == 2) ? 1 : ((grp == 3) ? -1 : 0);
        int dw = (grp == 0) ? 1 : ((grp == 1) ? -1 : 0);
        int oy = y0 + (pix >> 3) + dh - PADS;
        int ox = x0 + (pix & 7)  + dw - PADS;
        float v = 0.f;
        if (oy >= 0 && oy < HIMG && ox >= 0 && ox < HIMG)
            v = x[(((size_t)n * C_IN + i) * HIMG + oy) * HIMG + ox];
        sh_scr[pix * C_IN + i] = (f16)v;
    }
    __syncthreads();

    // ---- stage 1b: QKV GEMM. 3 convs x 6 M-tiles x 4 N-tiles = 72 tiles ----
    for (int t = wv; t < 72; t += 8) {
        int conv = t / 24;
        int rem  = t % 24;
        int mt = rem >> 2, nt = rem & 3;
        v8f acc = {0.f, 0.f, 0.f, 0.f, 0.f, 0.f, 0.f, 0.f};
        const f16* wf = wf_qkv + conv * (MIDC * C_IN);
#pragma unroll
        for (int kt = 0; kt < 6; ++kt) {
            int k0 = kt * 32 + lgr * 16;
            v16h A = frag_load(wf + (mt * 16 + lrow) * C_IN + k0);
            v16h B = frag_load(&sh_scr[(nt * 16 + lrow) * C_IN + k0]);
            acc = wmma_f16(A, B, acc);
        }
#pragma unroll
        for (int r = 0; r < 8; ++r) {
            int row = mt * 16 + r + lgr * 8;          // output channel
            int pix = nt * 16 + lrow;                 // window token
            float val = acc[r] + bf_qkv[conv * MIDC + row];
            if (conv == 0)      sh_theta[pix * MIDC + row] = (f16)val;
            else if (conv == 1) sh_phi  [pix * MIDC + row] = (f16)val;
            else                sh_v    [row * 64 + pix]   = (f16)val;
        }
    }
    __syncthreads();

    // ---- stage 2: attention, 4 phases x 2 heads (sim reuses sh_scr) ----
    for (int phase = 0; phase < 4; ++phase) {
        const int hl = wv >> 2;               // head slot 0/1 in sim buffer
        const int hh = phase * 2 + hl;        // global head id
        const int mt = wv & 3;                // this wave's row tile
        const int ch0 = hh * CPH;
        f16* sim = &sh_scr[hl * 4096];        // [64][64] fp16

        // sim = q k^T   (K = 12, zero-padded to 32)
        v16h Aq = mask_headdim(
            frag_load(&sh_theta[(mt * 16 + lrow) * MIDC + ch0 + lgr * 16]), lgr);
#pragma unroll
        for (int nt = 0; nt < 4; ++nt) {
            v16h Bk = mask_headdim(
                frag_load(&sh_phi[(nt * 16 + lrow) * MIDC + ch0 + lgr * 16]), lgr);
            v8f acc = {0.f, 0.f, 0.f, 0.f, 0.f, 0.f, 0.f, 0.f};
            acc = wmma_f16(Aq, Bk, acc);
#pragma unroll
            for (int r = 0; r < 8; ++r)
                sim[(mt * 16 + r + lgr * 8) * 64 + nt * 16 + lrow] = (f16)acc[r];
        }
        __syncthreads();

        // softmax over rows (128 rows total: 2 heads x 64)
        if (tid < 128) {
            f16* sr = &sh_scr[(tid >> 6) * 4096 + (tid & 63) * 64];
            float mx = -3.0e38f;
            for (int j = 0; j < 64; ++j) mx = fmaxf(mx, (float)sr[j]);
            float sum = 0.f;
            for (int j = 0; j < 64; ++j) sum += __expf((float)sr[j] - mx);
            float inv = 1.f / sum;
            for (int j = 0; j < 64; ++j)
                sr[j] = (f16)(__expf((float)sr[j] - mx) * inv);
        }
        __syncthreads();

        // out = P V   (M = 64 pix, N = 12 ch padded to 16, K = 64)
        v8f acc = {0.f, 0.f, 0.f, 0.f, 0.f, 0.f, 0.f, 0.f};
#pragma unroll
        for (int kt = 0; kt < 2; ++kt) {
            int k0 = kt * 32 + lgr * 16;
            v16h Ap = frag_load(&sim[(mt * 16 + lrow) * 64 + k0]);
            v16h Bv = (lrow < CPH) ? frag_load(&sh_v[(ch0 + lrow) * 64 + k0])
                                   : frag_zero();
            acc = wmma_f16(Ap, Bv, acc);
        }
        // stage to LDS in [local_ch][pix] layout for coalesced stores
        if (lrow < CPH) {
            int chl = hl * CPH + lrow;        // 0..23 within this phase
#pragma unroll
            for (int r = 0; r < 8; ++r) {
                int pix = mt * 16 + r + lgr * 8;
                sh_stage[chl * 64 + pix] = (f16)acc[r];
            }
        }
        __syncthreads();

        // coalesced 16B stores: 24 channels x 8 window rows
        if (tid < 24 * 8) {
            int chl = tid >> 3;               // 0..23
            int row = tid & 7;                // window row
            int ch  = phase * 24 + chl;       // global channel
            const uint4* src = reinterpret_cast<const uint4*>(&sh_stage[chl * 64 + row * 8]);
            uint4* dst = reinterpret_cast<uint4*>(
                attn_out + (((size_t)n * MIDC + ch) * HP + (y0 + row)) * HP + x0);
            *dst = *src;
        }
        __syncthreads();
    }
}

// ---------------- kernel 2: out shift-conv (96 -> 192) + crop ---------------
// One workgroup per 64 consecutive output pixels along W of one row.

__global__ __launch_bounds__(256)
void out_conv(const f16* __restrict__ attn_out,   // (4,96,264,264) fp16
              const f16* __restrict__ wf_out,     // (192,96) fp16
              const float* __restrict__ b_out,    // (192,)
              float* __restrict__ out)            // (4,192,256,256)
{
    __shared__ __align__(16) f16 sh_b[64 * MIDC];  // [pix][k]

    const int tid  = threadIdx.x;
    const int wv   = tid >> 5;
    const int lane = tid & 31;
    const int lrow = lane & 15;
    const int lgr  = lane >> 4;

    const int bx  = blockIdx.x;
    const int seg = bx & 3;               // 64-pixel segment of the row
    const int h   = (bx >> 2) & 255;      // output row
    const int n   = bx >> 10;             // batch

    // warm the fp16 weight stream into cache while we gather features
    __builtin_prefetch(wf_out + (tid << 6), 0, 3);

    // gather shifted attention features (always in padded bounds)
    for (int idx = tid; idx < 64 * MIDC; idx += 256) {
        int pix = idx & 63;
        int i   = idx >> 6;
        int grp = i / 19; if (grp > 4) grp = 4;
        int dh = (grp == 2) ? 1 : ((grp == 3) ? -1 : 0);
        int dw = (grp == 0) ? 1 : ((grp == 1) ? -1 : 0);
        int py = h + PADS + dh;
        int px = seg * 64 + pix + PADS + dw;
        sh_b[pix * MIDC + i] =
            attn_out[(((size_t)n * MIDC + i) * HP + py) * HP + px];
    }
    __syncthreads();

    // 12 M-tiles x 4 N-tiles = 48 tiles over 8 waves
    for (int t = wv; t < 48; t += 8) {
        int mt = t >> 2, nt = t & 3;
        v8f acc = {0.f, 0.f, 0.f, 0.f, 0.f, 0.f, 0.f, 0.f};
#pragma unroll
        for (int kt = 0; kt < 3; ++kt) {
            int k0 = kt * 32 + lgr * 16;
            v16h A = frag_load(wf_out + (mt * 16 + lrow) * MIDC + k0);
            v16h B = frag_load(&sh_b[(nt * 16 + lrow) * MIDC + k0]);
            acc = wmma_f16(A, B, acc);
        }
        int wcol = seg * 64 + nt * 16 + lrow;
#pragma unroll
        for (int r = 0; r < 8; ++r) {
            int row = mt * 16 + r + lgr * 8;   // output channel
            out[(((size_t)n * C_IN + row) * HIMG + h) * HIMG + wcol] =
                acc[r] + b_out[row];
        }
    }
}

// ---------------- host launcher ----------------

extern "C" void kernel_launch(void* const* d_in, const int* in_sizes, int n_in,
                              void* d_out, int out_size, void* d_ws, size_t ws_size,
                              hipStream_t stream) {
    const float* x = (const float*)d_in[0];
    FoldArgs fa;
    for (int c = 0; c < 3; ++c) {
        int base = 1 + c * 6;
        fa.w[c]     = (const float*)d_in[base + 0];
        fa.b[c]     = (const float*)d_in[base + 1];
        fa.gamma[c] = (const float*)d_in[base + 2];
        fa.beta[c]  = (const float*)d_in[base + 3];
        fa.mean[c]  = (const float*)d_in[base + 4];
        fa.var[c]   = (const float*)d_in[base + 5];
    }
    fa.w_out = (const float*)d_in[19];
    const float* b_out = (const float*)d_in[20];

    char* ws = (char*)d_ws;
    // workspace layout
    f16*   wf_qkv = (f16*)  (ws + 0);        // 3*96*192 fp16 = 110592 B
    float* bf_qkv = (float*)(ws + 110592);   // 288 f32       =   1152 B
    f16*   wf_out = (f16*)  (ws + 111744);   // 192*96 fp16   =  36864 B
    f16*   attn   = (f16*)  (ws + 148736);   // 4*96*264*264 fp16 = 53.5 MB

    fold_weights<<<128, 256, 0, stream>>>(fa, wf_qkv, bf_qkv, wf_out);

    const int nwin = 4 * NWINB;              // 4356 windows
    window_attn<<<nwin, 256, 0, stream>>>(x, wf_qkv, bf_qkv, attn);

    out_conv<<<4 * 256 * 4, 256, 0, stream>>>(attn, wf_out, b_out, (float*)d_out);
}
